// AttentionOffsetCorrelationLayer_12833362280730
// MI455X (gfx1250) — matrix operand order repaired and checked by reading, not compiled, not run on hardware
//
#include <hip/hip_runtime.h>

// CDNA5 / gfx1250: wave32, WMMA 16x16x32 bf16.
typedef __attribute__((ext_vector_type(16))) __bf16 v16bf;
typedef __attribute__((ext_vector_type(8)))  float  v8f;

#define Bq 2
#define Cc 256
#define Hh 80
#define Ww 160
#define Gg 4
#define Ss 9
#define HWp (Hh * Ww)
#define ROWP (Cc + 8)   // LDS row pad: 264 ushorts = 528B = 132 dwords -> bank = 4*m

__device__ __forceinline__ unsigned short f2bf(float f) {
  union { float f; unsigned u; } c; c.f = f;
  unsigned u = c.u;
  u += 0x7FFFu + ((u >> 16) & 1u);          // round-to-nearest-even
  return (unsigned short)(u >> 16);
}

__global__ __launch_bounds__(32)
void corr_wmma_kernel(const float* __restrict__ left,
                      const float* __restrict__ right,
                      const float* __restrict__ flow,
                      const float* __restrict__ extra,
                      float* __restrict__ out)
{
  // one wave (32 lanes) per output pixel
  const int pid  = blockIdx.x;
  const int w    = pid % Ww;
  const int h    = (pid / Ww) % Hh;
  const int b    = pid / (Ww * Hh);
  const int lane = threadIdx.x;

  // 16 rows so A rows 9..15 read genuine zeros (no divergent fragment fill)
  __shared__ __align__(16) unsigned short samp[16][ROWP]; // bf16 sampled (padded S x C)
  __shared__ __align__(16) unsigned short lft[ROWP];      // bf16 left (C)
  __shared__ __align__(16) unsigned short zb[16];         // 32B of zeros for B's dead columns

  // zero rows 9..15 of samp (contiguous 3696B = 231 uint4) and the zero block
  {
    uint4* z = (uint4*)&samp[Ss][0];
    const uint4 zero4 = make_uint4(0u, 0u, 0u, 0u);
    for (int i = lane; i < (16 - Ss) * ROWP / 8; i += 32) z[i] = zero4;
    if (lane < 8) ((unsigned short*)zb)[lane * 2] = 0, ((unsigned short*)zb)[lane * 2 + 1] = 0;
  }

  const float* lbase = left  + (size_t)b * Cc * HWp + (size_t)h * Ww + w;
  const float* rbase = right + (size_t)b * Cc * HWp;

  // stage left features as bf16 (lane = channel)
  #pragma unroll
  for (int cb = 0; cb < 8; ++cb) {
    const int c = cb * 32 + lane;
    lft[c] = f2bf(lbase[(size_t)c * HWp]);
  }

  const float fx = flow[((size_t)b * 2 + 0) * HWp + (size_t)h * Ww + w];
  const float fy = flow[((size_t)b * 2 + 1) * HWp + (size_t)h * Ww + w];

  // gather + bilinear-blend 9 x 256 samples into LDS as bf16
  for (int s = 0; s < Ss; ++s) {
    const float ex = extra[((size_t)b * (2 * Ss) + 2 * s + 0) * HWp + (size_t)h * Ww + w];
    const float ey = extra[((size_t)b * (2 * Ss) + 2 * s + 1) * HWp + (size_t)h * Ww + w];
    const float x  = (float)w + fx + (float)(s - 4) + ex;  // 1x9 window: off_x = s-4, off_y = 0
    const float y  = (float)h + fy + ey;
    const float x0f = floorf(x), y0f = floorf(y);
    const float wx = x - x0f, wy = y - y0f;
    const int x0 = (int)x0f, y0 = (int)y0f;
    const int x1 = x0 + 1,   y1 = y0 + 1;
    // zero-padding validity folded into the 4 (wave-uniform) corner weights
    const float vx0 = (x0 >= 0 && x0 < Ww) ? 1.f : 0.f;
    const float vx1 = (x1 >= 0 && x1 < Ww) ? 1.f : 0.f;
    const float vy0 = (y0 >= 0 && y0 < Hh) ? 1.f : 0.f;
    const float vy1 = (y1 >= 0 && y1 < Hh) ? 1.f : 0.f;
    const float w00 = (1.f - wx) * (1.f - wy) * vx0 * vy0;
    const float w01 = wx * (1.f - wy) * vx1 * vy0;
    const float w10 = (1.f - wx) * wy * vx0 * vy1;
    const float w11 = wx * wy * vx1 * vy1;
    const int cx0 = min(max(x0, 0), Ww - 1), cx1 = min(max(x1, 0), Ww - 1);
    const int cy0 = min(max(y0, 0), Hh - 1), cy1 = min(max(y1, 0), Hh - 1);
    const size_t i00 = (size_t)cy0 * Ww + cx0;
    const size_t i01 = (size_t)cy0 * Ww + cx1;
    const size_t i10 = (size_t)cy1 * Ww + cx0;
    const size_t i11 = (size_t)cy1 * Ww + cx1;
    #pragma unroll
    for (int cb = 0; cb < 8; ++cb) {
      const int c = cb * 32 + lane;                  // lane = channel
      const float* rc = rbase + (size_t)c * HWp;
      const float v = w00 * rc[i00] + w01 * rc[i01] + w10 * rc[i10] + w11 * rc[i11];
      samp[s][c] = f2bf(v);
    }
  }
  __syncthreads();

  // D[s][g] = sum_k samp[s][k] * left[k] * (k in group g), K = 256 in 8 chunks of 32.
  // A (16x32 bf16): lane L -> M=L&15, K = 16*hi .. 16*hi+15 contiguous, hi = L>>4.
  // B (32x16 bf16): lane = column n, K = 16*hi .. 16*hi+15 contiguous.
  // All lanes always load (full EXEC); dead lanes read zeroed LDS via address select.
  const int m  = lane & 15;
  const int hi = lane >> 4;
  const int kA = hi * 16;
  v8f acc = {};
  #pragma unroll
  for (int chunk = 0; chunk < 8; ++chunk) {
    const int k0 = chunk * 32;
    const int g0 = chunk >> 1;        // the group that owns channels [k0, k0+32)
    union { v16bf v; uint4 q[2]; } afrag, bfrag;
    const unsigned short* ap = &samp[m][k0 + kA];
    afrag.q[0] = *(const uint4*)ap;
    afrag.q[1] = *(const uint4*)(ap + 8);
    const unsigned short* bp = (m == g0) ? &lft[k0 + 16 * hi] : &zb[0];
    bfrag.q[0] = *(const uint4*)bp;
    bfrag.q[1] = *(const uint4*)(bp + 8);
    // (neg_a, A, neg_b, B, c_mod, C, reuse_a, reuse_b) -> v_wmma_f32_16x16x32_bf16
    acc = __builtin_amdgcn_wmma_f32_16x16x32_bf16(
        false, afrag.v, false, bfrag.v, (short)0, acc, false, false);
  }

  // D layout: lane holds column n = lane&15, VGPR r -> row M = r + 8*hi.
  // Valid outputs: n < 4 groups, M < 9 window positions. out channel = g*9 + s.
  const float scale = 1.0f / 64.0f;   // mean over 64 group channels
  if (m < Gg) {
    const int g = m;
    #pragma unroll
    for (int r = 0; r < 8; ++r) {
      const int sidx = r + 8 * hi;
      if (sidx < Ss) {
        out[(((size_t)b * (Gg * Ss) + g * Ss + sidx) * Hh + h) * Ww + w] = acc[r] * scale;
      }
    }
  }
}

extern "C" void kernel_launch(void* const* d_in, const int* in_sizes, int n_in,
                              void* d_out, int out_size, void* d_ws, size_t ws_size,
                              hipStream_t stream) {
  (void)in_sizes; (void)n_in; (void)out_size; (void)d_ws; (void)ws_size;
  const float* left  = (const float*)d_in[0];
  const float* right = (const float*)d_in[1];
  const float* flow  = (const float*)d_in[2];
  const float* extra = (const float*)d_in[3];
  float* out = (float*)d_out;
  const int nblocks = Bq * Hh * Ww;   // one wave per output pixel
  corr_wmma_kernel<<<nblocks, 32, 0, stream>>>(left, right, flow, extra, out);
}
